// TransformerEncoderLayer_81630148428077
// MI455X (gfx1250) — compile-verified, matrix operand order checked by reading, not run on hardware
//
#include <hip/hip_runtime.h>

// ---------------------------------------------------------------------------
// MI455X / gfx1250 implementation: mixed-precision (f16-in, f32-accum) WMMA
// GEMM backbone + small fused elementwise/reduction kernels.
// ---------------------------------------------------------------------------

typedef __attribute__((ext_vector_type(16))) _Float16     v16h;
typedef __attribute__((ext_vector_type(8)))  float        v8f;
typedef __attribute__((ext_vector_type(4)))  float        f32x4;
typedef __attribute__((ext_vector_type(4)))  unsigned int u32x4;

union FragU { v16h v; u32x4 q[2]; };

#define BM 128
#define BN 128
#define BK 32
#define LDK 40   // padded row stride (halves): 80 bytes, keeps 16B alignment

// Problem constants
#define kB  2
#define kS  1024
#define kD  512
#define kT  2048       // B*S
#define kNH 8
#define kHD 64
#define kNS 8
#define kF3 1536
#define kE  32
#define kF2 1024
#define kR  64
#define kTOPK 4

// ---------------------------------------------------------------------------
// Generic batched GEMM: C[z] = act( preScale * A[z]*B[z] + bias[z] ) (+= C)
//   A: [M,K] row-major fp32 (lda), batch stride sA
//   B: BT==0 -> [K,N] row-major (ldb=N-ish);  BT==1 -> [N,K] row-major (A*B^T)
//   C: [M,N] row-major fp32
// fp32 is converted to f16 while staging into LDS; WMMA f32_16x16x32_f16.
// act: 0 none, 1 silu, 2 sigmoid
// ---------------------------------------------------------------------------
__global__ __launch_bounds__(256)
void gemm_wmma_kernel(const float* __restrict__ A, long long sA, int lda,
                      const float* __restrict__ Bm, long long sB, int ldb,
                      float* __restrict__ C, long long sC, int ldc,
                      const float* __restrict__ bias, long long sBias,
                      int M, int N, int K, int BT, int act, float preScale,
                      int accum)
{
  __shared__ _Float16 lA[BM][LDK];
  __shared__ _Float16 lB[BN][LDK];

  const int tid = threadIdx.x;
  const int bz  = blockIdx.z;
  A  += (long long)bz * sA;
  Bm += (long long)bz * sB;
  C  += (long long)bz * sC;
  const float* biasp = bias ? bias + (long long)bz * sBias : nullptr;

  const int m0 = blockIdx.y * BM;
  const int n0 = blockIdx.x * BN;
  const int wave = tid >> 5, lane = tid & 31;
  const int wr = wave & 3, wc = wave >> 2;      // 4x2 wave grid, 32x64 per wave
  const int lrow = lane & 15;
  const int kb   = (lane >> 4) ? 8 : 0;          // k-half select per ISA layout

  const v8f vzero = {0.f,0.f,0.f,0.f,0.f,0.f,0.f,0.f};
  v8f acc[2][4];
#pragma unroll
  for (int i = 0; i < 2; ++i)
#pragma unroll
    for (int j = 0; j < 4; ++j) acc[i][j] = vzero;

  const int nk = (K + BK - 1) / BK;
  for (int kt = 0; kt < nk; ++kt) {
    const int k0 = kt * BK;
    // ---- stage A tile (128x32), coalesced float4, convert to f16 ----
#pragma unroll
    for (int i = 0; i < 4; ++i) {
      int r = (tid >> 3) + 32 * i;
      int c = (tid & 7) * 4;
      int gm = m0 + r, gk = k0 + c;
      f32x4 f = {0.f, 0.f, 0.f, 0.f};
      if (gm < M && gk < K) {
        f = *(const f32x4*)(A + (long long)gm * lda + gk);
        if (gk + BK < K)
          __builtin_prefetch(A + (long long)gm * lda + gk + BK, 0, 1);
      }
      _Float16* p = &lA[r][c];
      p[0] = (_Float16)f.x; p[1] = (_Float16)f.y;
      p[2] = (_Float16)f.z; p[3] = (_Float16)f.w;
    }
    // ---- stage B tile as [n][k] in LDS ----
    if (BT) {          // B is [N,K] row-major: straight coalesced copy
#pragma unroll
      for (int i = 0; i < 4; ++i) {
        int r = (tid >> 3) + 32 * i;
        int c = (tid & 7) * 4;
        int gn = n0 + r, gk = k0 + c;
        f32x4 f = {0.f, 0.f, 0.f, 0.f};
        if (gn < N && gk < K) {
          f = *(const f32x4*)(Bm + (long long)gn * ldb + gk);
          if (gk + BK < K)
            __builtin_prefetch(Bm + (long long)gn * ldb + gk + BK, 0, 1);
        }
        _Float16* p = &lB[r][c];
        p[0] = (_Float16)f.x; p[1] = (_Float16)f.y;
        p[2] = (_Float16)f.z; p[3] = (_Float16)f.w;
      }
    } else {           // B is [K,N]: coalesced along n, transpose into LDS
#pragma unroll
      for (int i = 0; i < 4; ++i) {
        int kk = (tid >> 5) + 8 * i;
        int nn = (tid & 31) * 4;
        int gk = k0 + kk, gn = n0 + nn;
        f32x4 f = {0.f, 0.f, 0.f, 0.f};
        if (gk < K && gn < N)
          f = *(const f32x4*)(Bm + (long long)gk * ldb + gn);
        lB[nn + 0][kk] = (_Float16)f.x;
        lB[nn + 1][kk] = (_Float16)f.y;
        lB[nn + 2][kk] = (_Float16)f.z;
        lB[nn + 3][kk] = (_Float16)f.w;
      }
    }
    __syncthreads();

    // ---- fragment gather (two ds_load_b128 per frag) + WMMA ----
    FragU a[2], b[4];
#pragma unroll
    for (int mi = 0; mi < 2; ++mi) {
      const _Float16* p = &lA[wr * 32 + mi * 16 + lrow][kb];
      a[mi].q[0] = *(const u32x4*)p;
      a[mi].q[1] = *(const u32x4*)(p + 16);
    }
#pragma unroll
    for (int ni = 0; ni < 4; ++ni) {
      const _Float16* p = &lB[wc * 64 + ni * 16 + lrow][kb];
      b[ni].q[0] = *(const u32x4*)p;
      b[ni].q[1] = *(const u32x4*)(p + 16);
    }
#pragma unroll
    for (int mi = 0; mi < 2; ++mi)
#pragma unroll
      for (int ni = 0; ni < 4; ++ni)
        acc[mi][ni] = __builtin_amdgcn_wmma_f32_16x16x32_f16(
            false, a[mi].v, false, b[ni].v, (short)0, acc[mi][ni],
            false, false);
    __syncthreads();
  }

  // ---- epilogue: C/D layout -> lane=(N within 16-tile), vgpr j -> M rows ----
  const int rshift = (lane >> 4) * 8;
#pragma unroll
  for (int mi = 0; mi < 2; ++mi) {
#pragma unroll
    for (int ni = 0; ni < 4; ++ni) {
      int n = n0 + wc * 64 + ni * 16 + lrow;
      if (n >= N) continue;
      float bv = biasp ? biasp[n] : 0.f;
      int mbase = m0 + wr * 32 + mi * 16 + rshift;
#pragma unroll
      for (int j = 0; j < 8; ++j) {
        int m = mbase + j;
        if (m >= M) continue;
        float vv = acc[mi][ni][j] * preScale + bv;
        if (act == 1)      vv = vv / (1.f + __expf(-vv));   // silu
        else if (act == 2) vv = 1.f / (1.f + __expf(-vv));  // sigmoid
        long long idx = (long long)m * ldc + n;
        if (accum) vv += C[idx];
        C[idx] = vv;
      }
    }
  }
}

// ---------------------------------------------------------------------------
// LayerNorm over D=512 (one block of 256 per row; 2 elems/thread)
// ---------------------------------------------------------------------------
__global__ __launch_bounds__(256)
void layernorm_kernel(const float* __restrict__ x, const float* __restrict__ w,
                      const float* __restrict__ b, float* __restrict__ y)
{
  __shared__ float red[8];
  const long long row = blockIdx.x;
  const float* p = x + row * kD;
  const int tid = threadIdx.x;
  float a0 = p[tid], a1 = p[tid + 256];
  float s = a0 + a1;
  for (int off = 16; off; off >>= 1) s += __shfl_xor(s, off);
  if ((tid & 31) == 0) red[tid >> 5] = s;
  __syncthreads();
  float tot = 0.f;
  for (int i = 0; i < 8; ++i) tot += red[i];
  const float mean = tot / (float)kD;
  __syncthreads();
  float d0 = a0 - mean, d1 = a1 - mean;
  s = d0 * d0 + d1 * d1;
  for (int off = 16; off; off >>= 1) s += __shfl_xor(s, off);
  if ((tid & 31) == 0) red[tid >> 5] = s;
  __syncthreads();
  tot = 0.f;
  for (int i = 0; i < 8; ++i) tot += red[i];
  const float inv = rsqrtf(tot / (float)kD + 1e-5f);
  y[row * kD + tid]       = d0 * inv * w[tid]       + b[tid];
  y[row * kD + tid + 256] = d1 * inv * w[tid + 256] + b[tid + 256];
}

// RMS-normalize rows of width 512 (no weight): rb = x * rsqrt(mean(x^2)+1e-6)
__global__ __launch_bounds__(256)
void rms512_kernel(const float* __restrict__ x, float* __restrict__ y)
{
  __shared__ float red[8];
  const long long row = blockIdx.x;
  const float* p = x + row * kD;
  const int tid = threadIdx.x;
  float a0 = p[tid], a1 = p[tid + 256];
  float s = a0 * a0 + a1 * a1;
  for (int off = 16; off; off >>= 1) s += __shfl_xor(s, off);
  if ((tid & 31) == 0) red[tid >> 5] = s;
  __syncthreads();
  float tot = 0.f;
  for (int i = 0; i < 8; ++i) tot += red[i];
  const float inv = rsqrtf(tot / (float)kD + 1e-6f);
  y[row * kD + tid]       = a0 * inv;
  y[row * kD + tid + 256] = a1 * inv;
}

// L2-normalize rows of width 64 (one wave per row, 8 rows per block)
__global__ __launch_bounds__(256)
void l2norm64_kernel(float* __restrict__ x, long long nrows)
{
  long long row = (long long)blockIdx.x * 8 + (threadIdx.x >> 5);
  if (row >= nrows) return;
  const int l = threadIdx.x & 31;
  float* p = x + row * 64;
  float a = p[l], b = p[l + 32];
  float ss = a * a + b * b;
  for (int off = 16; off; off >>= 1) ss += __shfl_xor(ss, off);
  float nrm = fmaxf(sqrtf(ss), 1e-12f);
  p[l]      = a / nrm;
  p[l + 32] = b / nrm;
}

// o = rms(o, arms_w[h]) * shortcut  over rows of width 64
__global__ __launch_bounds__(256)
void rms_shortcut_kernel(float* __restrict__ o, const float* __restrict__ sc,
                         const float* __restrict__ w, long long nrows,
                         long long rows_per_head)
{
  long long row = (long long)blockIdx.x * 8 + (threadIdx.x >> 5);
  if (row >= nrows) return;
  const int l = threadIdx.x & 31;
  float* p = o + row * 64;
  const float* s = sc + row * 64;
  const int h = (int)(row / rows_per_head);
  float a = p[l], b = p[l + 32];
  float ss = a * a + b * b;
  for (int off = 16; off; off >>= 1) ss += __shfl_xor(ss, off);
  const float inv = rsqrtf(ss / 64.f + 1e-6f);
  p[l]      = a * inv * w[h * 64 + l]      * s[l];
  p[l + 32] = b * inv * w[h * 64 + l + 32] * s[l + 32];
}

// Row softmax over 1024 cols (one 256-thread block per row, in-place)
__global__ __launch_bounds__(256)
void softmax1024_kernel(float* __restrict__ sdat)
{
  __shared__ float red[8];
  const long long row = blockIdx.x;
  float* p = sdat + row * (long long)kS;
  const int tid = threadIdx.x;
  float v0 = p[tid], v1 = p[tid + 256], v2 = p[tid + 512], v3 = p[tid + 768];
  float mx = fmaxf(fmaxf(v0, v1), fmaxf(v2, v3));
  for (int off = 16; off; off >>= 1) mx = fmaxf(mx, __shfl_xor(mx, off));
  if ((tid & 31) == 0) red[tid >> 5] = mx;
  __syncthreads();
  float m = red[0];
  for (int i = 1; i < 8; ++i) m = fmaxf(m, red[i]);
  __syncthreads();
  v0 = __expf(v0 - m); v1 = __expf(v1 - m);
  v2 = __expf(v2 - m); v3 = __expf(v3 - m);
  float s = v0 + v1 + v2 + v3;
  for (int off = 16; off; off >>= 1) s += __shfl_xor(s, off);
  if ((tid & 31) == 0) red[tid >> 5] = s;
  __syncthreads();
  float tot = 0.f;
  for (int i = 0; i < 8; ++i) tot += red[i];
  const float inv = 1.f / tot;
  p[tid] = v0 * inv; p[tid + 256] = v1 * inv;
  p[tid + 512] = v2 * inv; p[tid + 768] = v3 * inv;
}

// ---------------- elementwise kernels (grid-stride) ----------------
__global__ void ew_zero_kernel(float* p, long long n) {
  for (long long i = (long long)blockIdx.x * blockDim.x + threadIdx.x; i < n;
       i += (long long)gridDim.x * blockDim.x) p[i] = 0.f;
}
__global__ void ew_add_kernel(const float* a, const float* b, float* c, long long n) {
  for (long long i = (long long)blockIdx.x * blockDim.x + threadIdx.x; i < n;
       i += (long long)gridDim.x * blockDim.x) c[i] = a[i] + b[i];
}
__global__ void ew_mul_kernel(float* a, const float* b, long long n) {
  for (long long i = (long long)blockIdx.x * blockDim.x + threadIdx.x; i < n;
       i += (long long)gridDim.x * blockDim.x) a[i] *= b[i];
}
__global__ void ew_vgate_kernel(float* v, const float* al, const float* be, long long n) {
  for (long long i = (long long)blockIdx.x * blockDim.x + threadIdx.x; i < n;
       i += (long long)gridDim.x * blockDim.x) v[i] = v[i] * al[i] + be[i];
}
// hn = rb * w[d]  (row-broadcast of a 512-wide weight)
__global__ void rowscale_kernel(const float* rb, const float* w, float* hn, long long n) {
  for (long long i = (long long)blockIdx.x * blockDim.x + threadIdx.x; i < n;
       i += (long long)gridDim.x * blockDim.x) hn[i] = rb[i] * w[i & 511];
}
// routed += eo * w_full[t, e]
__global__ void axpyw_kernel(float* routed, const float* eo, const float* wf,
                             int e, long long n) {
  for (long long i = (long long)blockIdx.x * blockDim.x + threadIdx.x; i < n;
       i += (long long)gridDim.x * blockDim.x)
    routed[i] += eo[i] * wf[(i >> 9) * kE + e];
}
// cat[t][h*64+j] = o2[h][t][j]
__global__ void concat_heads_kernel(const float* o2, float* cat, long long n) {
  for (long long i = (long long)blockIdx.x * blockDim.x + threadIdx.x; i < n;
       i += (long long)gridDim.x * blockDim.x) {
    long long t = i >> 9;
    int c = (int)(i & 511);
    int h = c >> 6, j = c & 63;
    cat[i] = o2[((long long)h * kT + t) * 64 + j];
  }
}
// out = x1 + xf + G/NS + routed
__global__ void final_combine_kernel(const float* x1, const float* xf,
                                     const float* G, const float* routed,
                                     float* out, long long n) {
  for (long long i = (long long)blockIdx.x * blockDim.x + threadIdx.x; i < n;
       i += (long long)gridDim.x * blockDim.x)
    out[i] = x1[i] + xf[i] + G[i] * (1.f / (float)kNS) + routed[i];
}

// Router: per-token top-4 over 32 logits, softmax weights, expert counts
__global__ __launch_bounds__(256)
void router_topk_kernel(const float* __restrict__ logits,
                        float* __restrict__ wfull, float* __restrict__ counts)
{
  int t = blockIdx.x * 256 + threadIdx.x;
  if (t >= kT) return;
  const float* lg = logits + (long long)t * kE;
  bool used[kE];
#pragma unroll
  for (int e = 0; e < kE; ++e) used[e] = false;
  float tv[kTOPK]; int ti[kTOPK];
#pragma unroll
  for (int k = 0; k < kTOPK; ++k) {
    float best = -3.0e38f; int bi = 0;
    for (int e = 0; e < kE; ++e)
      if (!used[e] && lg[e] > best) { best = lg[e]; bi = e; }
    used[bi] = true; tv[k] = best; ti[k] = bi;
  }
  const float mx = tv[0];   // sorted descending
  float ev[kTOPK], s = 0.f;
#pragma unroll
  for (int k = 0; k < kTOPK; ++k) { ev[k] = __expf(tv[k] - mx); s += ev[k]; }
  float* wr = wfull + (long long)t * kE;
#pragma unroll
  for (int e = 0; e < kE; ++e) wr[e] = 0.f;
#pragma unroll
  for (int k = 0; k < kTOPK; ++k) {
    wr[ti[k]] += ev[k] / s;
    atomicAdd(&counts[ti[k]], 1.0f);
  }
}

// load_balance_loss = var(counts, ddof=1)
__global__ void lbl_kernel(const float* counts, float* out) {
  float m = 0.f;
  for (int e = 0; e < kE; ++e) m += counts[e];
  m /= (float)kE;
  float v = 0.f;
  for (int e = 0; e < kE; ++e) { float d = counts[e] - m; v += d * d; }
  out[0] = v / (float)(kE - 1);
}

// ---------------------------------------------------------------------------
// Host orchestration
// ---------------------------------------------------------------------------
extern "C" void kernel_launch(void* const* d_in, const int* in_sizes, int n_in,
                              void* d_out, int out_size, void* d_ws, size_t ws_size,
                              hipStream_t stream) {
  (void)in_sizes; (void)n_in; (void)out_size; (void)ws_size;
  const float* x      = (const float*)d_in[0];
  const float* ln1_w  = (const float*)d_in[1];
  const float* ln1_b  = (const float*)d_in[2];
  const float* ln2_w  = (const float*)d_in[3];
  const float* ln2_b  = (const float*)d_in[4];
  const float* q_w    = (const float*)d_in[5];
  const float* q_b    = (const float*)d_in[6];
  const float* k_w    = (const float*)d_in[7];
  const float* k_b    = (const float*)d_in[8];
  const float* v_w    = (const float*)d_in[9];
  const float* v_b    = (const float*)d_in[10];
  const float* aw1    = (const float*)d_in[11];
  const float* ab1    = (const float*)d_in[12];
  const float* aw2    = (const float*)d_in[13];
  const float* ab2    = (const float*)d_in[14];
  const float* bw     = (const float*)d_in[15];
  const float* bb     = (const float*)d_in[16];
  const float* scw1   = (const float*)d_in[17];
  const float* scb1   = (const float*)d_in[18];
  const float* scw2   = (const float*)d_in[19];
  const float* scb2   = (const float*)d_in[20];
  const float* arms_w = (const float*)d_in[21];
  const float* l1_w   = (const float*)d_in[22];
  const float* l1_b   = (const float*)d_in[23];
  const float* out_w  = (const float*)d_in[24];
  const float* out_b  = (const float*)d_in[25];
  const float* sh_rms = (const float*)d_in[26];
  const float* sh_w1  = (const float*)d_in[27];
  const float* sh_w2  = (const float*)d_in[28];
  const float* sh_w3  = (const float*)d_in[29];
  const float* r_w1   = (const float*)d_in[30];
  const float* r_w2   = (const float*)d_in[31];
  const float* r_w3   = (const float*)d_in[32];
  const float* rd_w   = (const float*)d_in[33];
  const float* ru_w   = (const float*)d_in[34];

  // ---- bump allocator over workspace (~135 MiB needed) ----
  char* wsp = (char*)d_ws;
  auto alloc = [&](long long nElems) -> float* {
    float* p = (float*)wsp;
    wsp += ((nElems * 4 + 255) / 256) * 256;
    return p;
  };
  float* xn     = alloc((long long)kT * kD);
  float* qb_    = alloc((long long)kNH * kT * kHD);
  float* kb_    = alloc((long long)kNH * kT * kHD);
  float* vb_    = alloc((long long)kNH * kT * kHD);
  float* alpha_ = alloc((long long)kNH * kT * kHD);
  float* beta_  = alloc((long long)kNH * kT * kHD);
  float* shct_  = alloc((long long)kNH * kT * kHD);
  float* ar1    = alloc((long long)kNH * kT * 32);
  float* scores = alloc((long long)kNH * kB * kS * kS);  // 16.8M floats
  float* o_     = alloc((long long)kNH * kT * kHD);
  float* o2_    = alloc((long long)kNH * kT * kHD);
  float* cat_   = alloc((long long)kT * kD);
  float* attn_  = alloc((long long)kT * kD);
  float* x1_    = alloc((long long)kT * kD);
  float* xf_    = alloc((long long)kT * kD);
  float* rb_    = alloc((long long)kT * kD);
  float* hn_    = alloc((long long)kT * kD);
  float* G_     = alloc((long long)kT * kD);
  float* lr_    = alloc((long long)kT * kR);
  float* logit_ = alloc((long long)kT * kE);
  float* wfull_ = alloc((long long)kT * kE);
  float* counts_= alloc(64);
  float* routed_= alloc((long long)kT * kD);
  // Big MoE temporaries alias the scores buffer (dead after attention):
  float* moe0 = scores;                          // h1 / g    (T*F3 max)
  float* moe1 = scores + (long long)kT * kF3;    // h3 / vv
  float* moe2 = scores + 2LL * kT * kF3;         // eo  (T*D)

  auto gemm = [&](const float* Ap, long long sA, int lda,
                  const float* Bp, long long sB, int ldb, int BT,
                  float* Cp, long long sC, int ldc,
                  const float* bp, long long sbias,
                  int M, int N, int Kd, int batch, int act, float preScale,
                  int accum) {
    dim3 g((unsigned)((N + BN - 1) / BN), (unsigned)((M + BM - 1) / BM),
           (unsigned)batch);
    gemm_wmma_kernel<<<g, 256, 0, stream>>>(Ap, sA, lda, Bp, sB, ldb, Cp, sC,
                                            ldc, bp, sbias, M, N, Kd, BT, act,
                                            preScale, accum);
  };
  auto ewgrid = [](long long n) { return dim3((unsigned)((n + 255) / 256)); };

  const long long sHW  = (long long)kD * kHD;   // per-head [D,HD] weight stride
  const long long sHT  = (long long)kT * kHD;   // per-head [T,HD] act stride
  const long long sAtt = (long long)kS * kHD;   // per-(h,b) [S,HD] stride

  // ================= attention sublayer =================
  layernorm_kernel<<<kT, 256, 0, stream>>>(x, ln1_w, ln1_b, xn);

  // q/k/v/beta projections (batched over heads via grid.z)
  gemm(xn, 0, kD, q_w, sHW, kHD, 0, qb_, sHT, kHD, q_b, kHD, kT, kHD, kD, kNH, 1, 1.f, 0);
  gemm(xn, 0, kD, k_w, sHW, kHD, 0, kb_, sHT, kHD, k_b, kHD, kT, kHD, kD, kNH, 1, 1.f, 0);
  gemm(xn, 0, kD, v_w, sHW, kHD, 0, vb_, sHT, kHD, v_b, kHD, kT, kHD, kD, kNH, 1, 1.f, 0);
  gemm(xn, 0, kD, bw,  sHW, kHD, 0, beta_, sHT, kHD, bb, kHD, kT, kHD, kD, kNH, 2, 1.f, 0);
  l2norm64_kernel<<<(kNH * kT) / 8, 256, 0, stream>>>(qb_, (long long)kNH * kT);
  l2norm64_kernel<<<(kNH * kT) / 8, 256, 0, stream>>>(kb_, (long long)kNH * kT);

  // alpha = sigmoid((xn@aw1 + ab1) @ aw2 + ab2)
  gemm(xn, 0, kD, aw1, (long long)kD * 32, 32, 0, ar1, (long long)kT * 32, 32,
       ab1, 32, kT, 32, kD, kNH, 0, 1.f, 0);
  gemm(ar1, (long long)kT * 32, 32, aw2, (long long)32 * kHD, kHD, 0, alpha_,
       sHT, kHD, ab2, kHD, kT, kHD, 32, kNH, 2, 1.f, 0);
  // shortcut = sigmoid((xn@scw1 + scb1) @ scw2 + scb2)   (reuses ar1)
  gemm(xn, 0, kD, scw1, (long long)kD * 32, 32, 0, ar1, (long long)kT * 32, 32,
       scb1, 32, kT, 32, kD, kNH, 0, 1.f, 0);
  gemm(ar1, (long long)kT * 32, 32, scw2, (long long)32 * kHD, kHD, 0, shct_,
       sHT, kHD, scb2, kHD, kT, kHD, 32, kNH, 2, 1.f, 0);

  // v = v*alpha + beta
  ew_vgate_kernel<<<ewgrid((long long)kNH * kT * kHD), 256, 0, stream>>>(
      vb_, alpha_, beta_, (long long)kNH * kT * kHD);

  // scores = q @ k^T / sqrt(HD); softmax; o = P @ v   (batch = NH*B)
  gemm(qb_, sAtt, kHD, kb_, sAtt, kHD, 1, scores, (long long)kS * kS, kS,
       nullptr, 0, kS, kS, kHD, kNH * kB, 0, 0.125f, 0);
  softmax1024_kernel<<<kNH * kB * kS, 256, 0, stream>>>(scores);
  gemm(scores, (long long)kS * kS, kS, vb_, sAtt, kHD, 0, o_, sAtt, kHD,
       nullptr, 0, kS, kHD, kS, kNH * kB, 0, 1.f, 0);

  // o = rms(o, arms_w) * shortcut; o2 = o @ l1_w + l1_b
  rms_shortcut_kernel<<<(kNH * kT) / 8, 256, 0, stream>>>(
      o_, shct_, arms_w, (long long)kNH * kT, (long long)kT);
  gemm(o_, sHT, kHD, l1_w, (long long)kHD * kHD, kHD, 0, o2_, sHT, kHD, l1_b,
       kHD, kT, kHD, kHD, kNH, 0, 1.f, 0);

  // concat heads -> [T,D]; output projection; residual
  concat_heads_kernel<<<ewgrid((long long)kT * kD), 256, 0, stream>>>(
      o2_, cat_, (long long)kT * kD);
  gemm(cat_, 0, kD, out_w, 0, kD, 0, attn_, 0, kD, out_b, 0, kT, kD, kD, 1, 0,
       1.f, 0);
  ew_add_kernel<<<ewgrid((long long)kT * kD), 256, 0, stream>>>(
      x, attn_, x1_, (long long)kT * kD);

  // ================= MoE sublayer =================
  layernorm_kernel<<<kT, 256, 0, stream>>>(x1_, ln2_w, ln2_b, xf_);
  rms512_kernel<<<kT, 256, 0, stream>>>(xf_, rb_);

  // shared experts: G += (silu(hn@w1) * (hn@w3)) @ w2  over NS experts
  ew_zero_kernel<<<ewgrid((long long)kT * kD), 256, 0, stream>>>(
      G_, (long long)kT * kD);
  for (int n = 0; n < kNS; ++n) {
    rowscale_kernel<<<ewgrid((long long)kT * kD), 256, 0, stream>>>(
        rb_, sh_rms + (long long)n * kD, hn_, (long long)kT * kD);
    gemm(hn_, 0, kD, sh_w1 + (long long)n * kD * kF3, 0, kF3, 0, moe0, 0, kF3,
         nullptr, 0, kT, kF3, kD, 1, 1, 1.f, 0);
    gemm(hn_, 0, kD, sh_w3 + (long long)n * kD * kF3, 0, kF3, 0, moe1, 0, kF3,
         nullptr, 0, kT, kF3, kD, 1, 0, 1.f, 0);
    ew_mul_kernel<<<ewgrid((long long)kT * kF3), 256, 0, stream>>>(
        moe0, moe1, (long long)kT * kF3);
    gemm(moe0, 0, kF3, sh_w2 + (long long)n * kF3 * kD, 0, kD, 0, G_, 0, kD,
         nullptr, 0, kT, kD, kF3, 1, 0, 1.f, 1);
  }

  // router: logits = (xf @ rd_w) @ ru_w; top-4 softmax
  gemm(xf_, 0, kD, rd_w, 0, kR, 0, lr_, 0, kR, nullptr, 0, kT, kR, kD, 1, 0, 1.f, 0);
  gemm(lr_, 0, kR, ru_w, 0, kE, 0, logit_, 0, kE, nullptr, 0, kT, kE, kR, 1, 0, 1.f, 0);
  ew_zero_kernel<<<1, 64, 0, stream>>>(counts_, kE);
  router_topk_kernel<<<(kT + 255) / 256, 256, 0, stream>>>(logit_, wfull_, counts_);

  // routed experts (dense math, per-expert weighted accumulate)
  ew_zero_kernel<<<ewgrid((long long)kT * kD), 256, 0, stream>>>(
      routed_, (long long)kT * kD);
  for (int e = 0; e < kE; ++e) {
    gemm(xf_, 0, kD, r_w1 + (long long)e * kD * kF2, 0, kF2, 0, moe0, 0, kF2,
         nullptr, 0, kT, kF2, kD, 1, 1, 1.f, 0);
    gemm(xf_, 0, kD, r_w3 + (long long)e * kD * kF2, 0, kF2, 0, moe1, 0, kF2,
         nullptr, 0, kT, kF2, kD, 1, 0, 1.f, 0);
    ew_mul_kernel<<<ewgrid((long long)kT * kF2), 256, 0, stream>>>(
        moe0, moe1, (long long)kT * kF2);
    gemm(moe0, 0, kF2, r_w2 + (long long)e * kF2 * kD, 0, kD, 0, moe2, 0, kD,
         nullptr, 0, kT, kD, kF2, 1, 0, 1.f, 0);
    axpyw_kernel<<<ewgrid((long long)kT * kD), 256, 0, stream>>>(
        routed_, moe2, wfull_, e, (long long)kT * kD);
  }

  // final: out = x1 + xf + G/NS + routed ; out[T*D] = load-balance loss
  float* out = (float*)d_out;
  final_combine_kernel<<<ewgrid((long long)kT * kD), 256, 0, stream>>>(
      x1_, xf_, G_, routed_, out, (long long)kT * kD);
  lbl_kernel<<<1, 1, 0, stream>>>(counts_, out + (long long)kT * kD);
}
